// RGAT_67078799229116
// MI455X (gfx1250) — matrix operand order repaired and checked by reading, not compiled, not run on hardware
//
#include <hip/hip_runtime.h>
#include <hip/hip_bf16.h>

// ---------------------------------------------------------------------------
// RGAT (2-layer relational GAT + link-pred decoder) for MI455X / gfx1250.
//
//  * Per-relation transforms -> v_wmma_f32_16x16x32_bf16 (bf16 A/B, f32 acc).
//  * Weights pre-swizzled once into WMMA B-fragment order: every B fragment
//    is two 16B global_load_b128-class loads.
//  * Each wave computes 4 n-tiles (TPW=4) reusing its A fragment in registers:
//    4x less A re-read traffic, 4 WMMAs per A load.
//  * Edge phase: 3 memory-bound passes with 16B loads + f32/u32 atomics.
// ---------------------------------------------------------------------------

#define NN    20000
#define EE    320000
#define RR    16
#define INF_  128
#define HIDF  64
#define HEADS 8
#define OUTF  128
#define TT    50000
#define F1    (HEADS * HIDF)   // 512

typedef __attribute__((ext_vector_type(8)))  __bf16 v8bf;
typedef __attribute__((ext_vector_type(16))) __bf16 v16bf;
typedef __attribute__((ext_vector_type(8)))  float  v8f;

// ---- ordered-uint encoding for float atomic max --------------------------
__device__ __forceinline__ unsigned fkey(float f) {
    unsigned u = __float_as_uint(f);
    return (u & 0x80000000u) ? ~u : (u | 0x80000000u);
}
__device__ __forceinline__ float unfkey(unsigned u) {
    unsigned b = (u & 0x80000000u) ? (u & 0x7fffffffu) : ~u;
    return __uint_as_float(b);
}

__device__ __forceinline__ v16bf cat8(v8bf lo, v8bf hi) {
    return __builtin_shufflevector(lo, hi, 0,1,2,3,4,5,6,7,8,9,10,11,12,13,14,15);
}

// ---- WMMA A fragment (CDNA5 ISA 7.12.2 16-bit A 16x32 layout) ------------
// lane 0-15: M=lane, K in {0..7,16..23}; lane 16-31: M=lane-16, K in {8..15,24..31}
__device__ __forceinline__ v16bf load_a_frag(const __bf16* __restrict__ row,
                                             int akb /* 0 or 8 */) {
    v8bf lo = *reinterpret_cast<const v8bf*>(row + akb);
    v8bf hi = *reinterpret_cast<const v8bf*>(row + akb + 16);
    return cat8(lo, hi);
}
// ---- WMMA B fragment from pre-swizzled weights: contiguous 32B per lane --
__device__ __forceinline__ v16bf load_b_frag_sw(const __bf16* __restrict__ blk,
                                                int lane) {
    const __bf16* p = blk + lane * 16;
    v8bf lo = *reinterpret_cast<const v8bf*>(p);
    v8bf hi = *reinterpret_cast<const v8bf*>(p + 8);
    return cat8(lo, hi);
}

// ---------------------------------------------------------------------------
// Weight pre-swizzle (fused f32->bf16): W[R,K,Nc] row-major ->
// Bt blocks: (((r*NT + nt)*KB + kb)*32 + lane)*16 + i  holds
//   W[r][kb*32 + (i>>3)*16 + (lane>>4)*8 + (i&7)][nt*16 + (lane&15)]
// ---------------------------------------------------------------------------
__global__ void swizzle_b(const float* __restrict__ W, __bf16* __restrict__ Bt,
                          int K, int Nc, int R) {
    const int NT = Nc / 16, KB = K / 32;
    size_t idx = blockIdx.x * (size_t)blockDim.x + threadIdx.x;
    size_t total = (size_t)R * K * Nc;
    if (idx >= total) return;
    const int i    = (int)(idx & 15);
    const int lane = (int)((idx >> 4) & 31);
    size_t rest = idx >> 9;
    const int kb = (int)(rest % KB); rest /= KB;
    const int nt = (int)(rest % NT);
    const int r  = (int)(rest / NT);
    const int krow = kb * 32 + (i >> 3) * 16 + (lane >> 4) * 8 + (i & 7);
    const int col  = nt * 16 + (lane & 15);
    Bt[idx] = (__bf16)W[((size_t)r * K + krow) * Nc + col];
}

// ---------------------------------------------------------------------------
// GEMM: C[r] = A @ B[r].  A:[M,K] bf16 row-major, Bt swizzled, C:[R,M,Nc] bf16.
// Each wave: one 16-row m-tile x TPW consecutive 16-col n-tiles, A fragment
// reused across TPW WMMAs per k-step. 8 waves/block.
// ---------------------------------------------------------------------------
template<int K, int Nc, int TPW>
__global__ void gemm_bf16_rel(const __bf16* __restrict__ A,
                              const __bf16* __restrict__ Bt,
                              __bf16* __restrict__ C, int M) {
    constexpr int NT  = Nc / 16, KB = K / 32;
    constexpr int WPM = NT / TPW;        // waves per m-tile
    constexpr int MPB = 8 / WPM;         // m-tiles per block
    const int lane = threadIdx.x & 31;
    const int wave = threadIdx.x >> 5;
    const int mt  = blockIdx.x * MPB + wave / WPM;
    const int nt0 = (wave % WPM) * TPW;
    const int r   = blockIdx.z;
    if (mt * 16 >= M) return;

    const int am  = lane & 15;
    const int akb = (lane >> 4) * 8;
    const __bf16* arow  = A + ((size_t)mt * 16 + am) * K;
    const __bf16* bbase = Bt + ((size_t)r * NT + nt0) * KB * 512;

    v8f acc[TPW];
#pragma unroll
    for (int j = 0; j < TPW; ++j) acc[j] = (v8f){};

#pragma unroll
    for (int kb = 0; kb < KB; ++kb) {
        v16bf a = load_a_frag(arow + kb * 32, akb);
#pragma unroll
        for (int j = 0; j < TPW; ++j) {
            v16bf b = load_b_frag_sw(bbase + ((size_t)j * KB + kb) * 512, lane);
            acc[j] = __builtin_amdgcn_wmma_f32_16x16x32_bf16(
                false, a, false, b, (short)0, acc[j], false, false);
        }
    }
    const int mb = (lane >> 4) * 8;
    const int cn = lane & 15;
    __bf16* Cr = C + (size_t)r * M * Nc + (size_t)mt * 16 * Nc + nt0 * 16 + cn;
#pragma unroll
    for (int j = 0; j < TPW; ++j)
#pragma unroll
        for (int v = 0; v < 8; ++v)
            Cr[(size_t)(mb + v) * Nc + j * 16] = (__bf16)acc[j][v];
}

// ---------------------------------------------------------------------------
// Decoder GEMM: hdec = relu(concat(z[te0],z[te1]) @ lw1 + lb1), bf16 out.
// A-row gather fused; TPW=4 n-tiles per wave (NT=8 -> 2 waves/m-tile).
// ---------------------------------------------------------------------------
__global__ void gemm_dec(const __bf16* __restrict__ z,
                         const int* __restrict__ te0,
                         const int* __restrict__ te1,
                         const __bf16* __restrict__ Bt,   // swizzled [256,128]
                         const float* __restrict__ bias,  // [128]
                         __bf16* __restrict__ Cd) {       // [T,128] bf16
    constexpr int NT = OUTF / 16, KB = 256 / 32, TPW = 4;
    constexpr int WPM = NT / TPW;        // 2 waves per m-tile
    constexpr int MPB = 8 / WPM;         // 4 m-tiles per block
    const int lane = threadIdx.x & 31;
    const int wave = threadIdx.x >> 5;
    const int mt  = blockIdx.x * MPB + wave / WPM;
    const int nt0 = (wave % WPM) * TPW;
    if (mt * 16 >= TT) return;

    const int am  = lane & 15;
    const int akb = (lane >> 4) * 8;
    const int t   = mt * 16 + am;
    const int i0  = te0[t], i1 = te1[t];
    const __bf16* bbase = Bt + (size_t)nt0 * KB * 512;

    v8f acc[TPW];
#pragma unroll
    for (int j = 0; j < TPW; ++j) acc[j] = (v8f){};

#pragma unroll
    for (int kb = 0; kb < KB; ++kb) {
        const int k0 = kb * 32;
        const __bf16* arow = z + (size_t)(k0 < 128 ? i0 : i1) * OUTF + (k0 & 127);
        v16bf a = load_a_frag(arow, akb);
#pragma unroll
        for (int j = 0; j < TPW; ++j) {
            v16bf b = load_b_frag_sw(bbase + ((size_t)j * KB + kb) * 512, lane);
            acc[j] = __builtin_amdgcn_wmma_f32_16x16x32_bf16(
                false, a, false, b, (short)0, acc[j], false, false);
        }
    }
    const int mb = (lane >> 4) * 8;
#pragma unroll
    for (int j = 0; j < TPW; ++j) {
        const int col = (nt0 + j) * 16 + (lane & 15);
        const float bi = bias[col];
#pragma unroll
        for (int v = 0; v < 8; ++v) {
            float f = acc[j][v] + bi;
            Cd[(size_t)(mt * 16 + mb + v) * OUTF + col] = (__bf16)(f > 0.f ? f : 0.f);
        }
    }
}

// ---------------------------------------------------------------------------
// Edge pass 1: logits = leaky_relu(q.aq + k.ak); atomic segment-max by dst.
// ---------------------------------------------------------------------------
__global__ void edge_logits(const __bf16* __restrict__ xr,
                            const int* __restrict__ src,
                            const int* __restrict__ dst,
                            const int* __restrict__ etype,
                            const float* __restrict__ attq,
                            const float* __restrict__ attk,
                            float* __restrict__ logits,
                            unsigned* __restrict__ mkey,
                            int Etot, int F, int O, int sh) {
    const int gid = blockIdx.x * blockDim.x + threadIdx.x;
    const int H = 1 << sh;
    if (gid >= Etot * H) return;
    const int e = gid >> sh, h = gid & (H - 1);
    const int et = etype[e], s = src[e], d = dst[e];
    const __bf16* kq = xr + ((size_t)et * NN + s) * F + h * O;
    const __bf16* qq = xr + ((size_t)et * NN + d) * F + h * O;
    const float* aq = attq + ((size_t)et * H + h) * O;
    const float* ak = attk + ((size_t)et * H + h) * O;
    float acc = 0.f;
    for (int o = 0; o < O; o += 8) {
        v8bf qv = *reinterpret_cast<const v8bf*>(qq + o);
        v8bf kv = *reinterpret_cast<const v8bf*>(kq + o);
#pragma unroll
        for (int j = 0; j < 8; ++j)
            acc += (float)qv[j] * aq[o + j] + (float)kv[j] * ak[o + j];
    }
    const float lg = acc > 0.f ? acc : 0.2f * acc;   // leaky_relu 0.2
    logits[gid] = lg;
    atomicMax(&mkey[(size_t)d * H + h], fkey(lg));
}

// ---------------------------------------------------------------------------
// Edge pass 2: e = exp(logit - m[dst]); atomic segment-sum by dst.
// ---------------------------------------------------------------------------
__global__ void edge_exp(float* __restrict__ logits,
                         const unsigned* __restrict__ mkey,
                         float* __restrict__ ssum,
                         const int* __restrict__ dst,
                         int Etot, int sh) {
    const int gid = blockIdx.x * blockDim.x + threadIdx.x;
    const int H = 1 << sh;
    if (gid >= Etot * H) return;
    const int e = gid >> sh, h = gid & (H - 1);
    const int d = dst[e];
    const float m = unfkey(mkey[(size_t)d * H + h]);
    const float ex = __expf(logits[gid] - m);
    logits[gid] = ex;
    atomicAdd(&ssum[(size_t)d * H + h], ex);
}

// ---------------------------------------------------------------------------
// Edge pass 3: out[dst] += (e / (s+eps)) * k   (scatter-add messages)
// ---------------------------------------------------------------------------
__global__ void edge_scatter(const __bf16* __restrict__ xr,
                             const float* __restrict__ ebuf,
                             const float* __restrict__ ssum,
                             const int* __restrict__ src,
                             const int* __restrict__ dst,
                             const int* __restrict__ etype,
                             float* __restrict__ out,
                             int Etot, int F, int O, int sh) {
    const int gid = blockIdx.x * blockDim.x + threadIdx.x;
    const int H = 1 << sh;
    if (gid >= Etot * H) return;
    const int e = gid >> sh, h = gid & (H - 1);
    const int et = etype[e], s = src[e], d = dst[e];
    const float a = ebuf[gid] / (ssum[(size_t)d * H + h] + 1e-16f);
    const __bf16* kq = xr + ((size_t)et * NN + s) * F + h * O;
    float* op = out + (size_t)d * F + h * O;
    for (int o = 0; o < O; o += 8) {
        v8bf kv = *reinterpret_cast<const v8bf*>(kq + o);
#pragma unroll
        for (int j = 0; j < 8; ++j)
            atomicAdd(&op[o + j], a * (float)kv[j]);
    }
}

// ---- f32 -> bf16 conversion (optional ReLU) ------------------------------
__global__ void cvt_bf16(const float* __restrict__ in, __bf16* __restrict__ out,
                         size_t n, int relu) {
    size_t i = blockIdx.x * (size_t)blockDim.x + threadIdx.x;
    if (i >= n) return;
    float v = in[i];
    if (relu && v < 0.f) v = 0.f;
    out[i] = (__bf16)v;
}

// ---- final decode: out[t] = relu_hdec[t] . lw2 + lb2 ---------------------
__global__ void decode_out(const __bf16* __restrict__ hdec,
                           const float* __restrict__ lw2,
                           const float* __restrict__ lb2,
                           float* __restrict__ out, int T) {
    const int t = blockIdx.x * blockDim.x + threadIdx.x;
    if (t >= T) return;
    const __bf16* hp = hdec + (size_t)t * OUTF;
    float acc = 0.f;
    for (int o = 0; o < OUTF; o += 8) {
        v8bf hv = *reinterpret_cast<const v8bf*>(hp + o);
#pragma unroll
        for (int j = 0; j < 8; ++j) acc += (float)hv[j] * lw2[o + j];
    }
    out[t] = acc + lb2[0];
}

// ---------------------------------------------------------------------------
extern "C" void kernel_launch(void* const* d_in, const int* in_sizes, int n_in,
                              void* d_out, int out_size, void* d_ws, size_t ws_size,
                              hipStream_t stream) {
    const float* x    = (const float*)d_in[0];
    const int*   ei   = (const int*)d_in[1];
    const int*   esrc = ei;
    const int*   edst = ei + EE;
    const int*   et   = (const int*)d_in[2];
    const int*   te   = (const int*)d_in[3];
    const int*   te0  = te;
    const int*   te1  = te + TT;
    const float* W1   = (const float*)d_in[4];
    const float* aq1  = (const float*)d_in[5];
    const float* ak1  = (const float*)d_in[6];
    const float* W2   = (const float*)d_in[7];
    const float* aq2  = (const float*)d_in[8];
    const float* ak2  = (const float*)d_in[9];
    const float* lw1  = (const float*)d_in[10];
    const float* lb1  = (const float*)d_in[11];
    const float* lw2  = (const float*)d_in[12];
    const float* lb2  = (const float*)d_in[13];
    float* outp = (float*)d_out;

    // ---- bump allocator over d_ws ----
    char* w = (char*)d_ws;
    auto alloc = [&](size_t bytes) -> void* {
        void* p = (void*)w;
        w += (bytes + 255) & ~(size_t)255;
        return p;
    };
    __bf16*   xbf   = (__bf16*)alloc((size_t)NN * INF_ * 2);
    __bf16*   w1sw  = (__bf16*)alloc((size_t)RR * INF_ * F1 * 2);
    __bf16*   w2sw  = (__bf16*)alloc((size_t)RR * F1 * OUTF * 2);
    __bf16*   lw1sw = (__bf16*)alloc((size_t)2 * OUTF * OUTF * 2);
    __bf16*   xr    = (__bf16*)alloc((size_t)RR * NN * F1 * 2);   // reused layer2
    float*    logit = (float*)alloc((size_t)EE * HEADS * 4);      // reused layer2
    unsigned* mkey  = (unsigned*)alloc((size_t)NN * HEADS * 4);
    float*    ssum  = (float*)alloc((size_t)NN * HEADS * 4);
    float*    accb  = (float*)alloc((size_t)NN * F1 * 4);         // reused layer2
    __bf16*   hbf   = (__bf16*)alloc((size_t)NN * F1 * 2);
    __bf16*   zbf   = (__bf16*)alloc((size_t)NN * OUTF * 2);
    __bf16*   hdec  = (__bf16*)alloc((size_t)TT * OUTF * 2);

    const int TB = 256;
    auto nb = [](size_t n) { return (unsigned)((n + 255) / 256); };

    // ---- staging: bf16 features + fragment-swizzled bf16 weights ----
    cvt_bf16<<<nb((size_t)NN * INF_), TB, 0, stream>>>(x, xbf, (size_t)NN * INF_, 0);
    swizzle_b<<<nb((size_t)RR * INF_ * F1), TB, 0, stream>>>(W1, w1sw, INF_, F1, RR);
    swizzle_b<<<nb((size_t)RR * F1 * OUTF), TB, 0, stream>>>(W2, w2sw, F1, OUTF, RR);
    swizzle_b<<<nb((size_t)2 * OUTF * OUTF), TB, 0, stream>>>(lw1, lw1sw, 2 * OUTF, OUTF, 1);

    // ================= layer 1 (HEADS=8, O=64, F=512) =================
    hipMemsetAsync(mkey, 0, (size_t)NN * HEADS * 4, stream);
    hipMemsetAsync(ssum, 0, (size_t)NN * HEADS * 4, stream);
    hipMemsetAsync(accb, 0, (size_t)NN * F1 * 4, stream);

    // NT=32, TPW=4 -> WPM=8, 1 m-tile per block
    gemm_bf16_rel<INF_, F1, 4><<<dim3(NN / 16, 1, RR), TB, 0, stream>>>(
        xbf, w1sw, xr, NN);

    edge_logits<<<nb((size_t)EE * HEADS), TB, 0, stream>>>(
        xr, esrc, edst, et, aq1, ak1, logit, mkey, EE, F1, HIDF, 3);
    edge_exp<<<nb((size_t)EE * HEADS), TB, 0, stream>>>(
        logit, mkey, ssum, edst, EE, 3);
    edge_scatter<<<nb((size_t)EE * HEADS), TB, 0, stream>>>(
        xr, logit, ssum, esrc, edst, et, accb, EE, F1, HIDF, 3);

    cvt_bf16<<<nb((size_t)NN * F1), TB, 0, stream>>>(accb, hbf, (size_t)NN * F1, 1); // relu

    // ================= layer 2 (HEADS=1, O=128, F=128) =================
    hipMemsetAsync(mkey, 0, (size_t)NN * 4, stream);
    hipMemsetAsync(ssum, 0, (size_t)NN * 4, stream);
    hipMemsetAsync(accb, 0, (size_t)NN * OUTF * 4, stream);

    // NT=8, TPW=4 -> WPM=2, 4 m-tiles per block; 1250 m-tiles -> 313 blocks
    gemm_bf16_rel<F1, OUTF, 4><<<dim3((NN / 16 + 3) / 4, 1, RR), TB, 0, stream>>>(
        hbf, w2sw, xr, NN);

    edge_logits<<<nb((size_t)EE), TB, 0, stream>>>(
        xr, esrc, edst, et, aq2, ak2, logit, mkey, EE, OUTF, OUTF, 0);
    edge_exp<<<nb((size_t)EE), TB, 0, stream>>>(
        logit, mkey, ssum, edst, EE, 0);
    edge_scatter<<<nb((size_t)EE), TB, 0, stream>>>(
        xr, logit, ssum, esrc, edst, et, accb, EE, OUTF, OUTF, 0);

    cvt_bf16<<<nb((size_t)NN * OUTF), TB, 0, stream>>>(accb, zbf, (size_t)NN * OUTF, 0);

    // ================= decoder =================
    // 3125 m-tiles, 4 per block -> 782 blocks
    gemm_dec<<<dim3((TT / 16 + 3) / 4, 1, 1), TB, 0, stream>>>(
        zbf, te0, te1, lw1sw, lb1, hdec);
    decode_out<<<nb((size_t)TT), TB, 0, stream>>>(hdec, lw2, lb2, outp, TT);

    (void)in_sizes; (void)n_in; (void)out_size; (void)ws_size;
}
